// MPNNLayer_27822798143740
// MI455X (gfx1250) — matrix operand order repaired and checked by reading, not compiled, not run on hardware
//
#include <hip/hip_runtime.h>
#include <hip/hip_bf16.h>
#include <math.h>

// GATv2 MPNN layer for MI455X (gfx1250, wave32, WMMA).
// Dense GEMMs use v_wmma_f32_16x16x32_bf16; irregular edge work uses native
// f32/u32 global atomics. Workspace layout (floats): xl[N*64] xr[N*64]
// exlog[E*4] maxb[N*4](uint) denom[N*4] stats[128]  (~40 MB).

typedef __attribute__((ext_vector_type(16))) __bf16 v16bf;
typedef __attribute__((ext_vector_type(8)))  __bf16 v8bf;
typedef __attribute__((ext_vector_type(8)))  float  v8f;

#define TPB 256

// monotonic unsigned encoding of float (for atomicMax-based segment max)
__device__ __forceinline__ unsigned ordf(float f) {
  unsigned u = __float_as_uint(f);
  return (u & 0x80000000u) ? ~u : (u | 0x80000000u);
}
__device__ __forceinline__ float unordf(unsigned u) {
  return __uint_as_float((u & 0x80000000u) ? (u & 0x7FFFFFFFu) : ~u);
}

// ---------------- kernel 0: init -------------------------------------------
__global__ __launch_bounds__(TPB) void k0_init(float* __restrict__ out,
                                               unsigned* __restrict__ maxb,
                                               float* __restrict__ denom,
                                               float* __restrict__ stats, int n) {
  int gid = blockIdx.x * TPB + threadIdx.x;
  if (gid < n * 64) out[gid] = 0.0f;
  if (gid < n * 4) { maxb[gid] = 0u; denom[gid] = 0.0f; }
  if (gid < 128) stats[gid] = 0.0f;
}

// ---------------- kernel 1: node transforms xl = xWl+bl, xr = xWr+br --------
// 8 waves/block, each wave owns a 16-row tile, loops over 8 column tiles
// (4 for W_l, 4 for W_r), K=64 as two k-steps of 32.  16 WMMAs / wave.
__global__ __launch_bounds__(TPB) void k1_node_xform(
    const float* __restrict__ x,
    const float* __restrict__ Wl, const float* __restrict__ bl,
    const float* __restrict__ Wr, const float* __restrict__ br,
    float* __restrict__ xl, float* __restrict__ xr, int n) {
  __shared__ __attribute__((aligned(16))) __bf16 sW[2 * 64 * 64]; // [sel][col][k] transposed
  int tid = threadIdx.x;
  for (int idx = tid; idx < 8192; idx += TPB) {
    int sel = idx >> 12;
    int rem = idx & 4095;
    int col = rem >> 6, k = rem & 63;
    const float* W = sel ? Wr : Wl;           // W stored row-major [k][col]
    sW[idx] = (__bf16)W[k * 64 + col];        // store transposed [col][k]
  }
  __syncthreads();

  int wave = tid >> 5, lane = tid & 31;
  int half = lane >> 4, l = lane & 15;
  int rbase = (blockIdx.x * 8 + wave) * 16;

  int row = rbase + l;
  int rowc = row < n ? row : (n - 1);
  const float* xp = x + (size_t)rowc * 64;

  // A fragments (16x32 bf16 layout): lane l holds row M=l.
  // half=0: elems 0..7 = K[s..s+8), elems 8..15 = K[s+16..s+24), s = ks*32
  // half=1: same with s += 8
  v16bf afrag[2];
#pragma unroll
  for (int ks = 0; ks < 2; ++ks) {
    int s1 = ks * 32 + 8 * half;
    int s2 = s1 + 16;
    float4 a0 = *(const float4*)(xp + s1);
    float4 a1 = *(const float4*)(xp + s1 + 4);
    float4 a2 = *(const float4*)(xp + s2);
    float4 a3 = *(const float4*)(xp + s2 + 4);
    v16bf a;
    a[0] = (__bf16)a0.x; a[1] = (__bf16)a0.y; a[2]  = (__bf16)a0.z; a[3]  = (__bf16)a0.w;
    a[4] = (__bf16)a1.x; a[5] = (__bf16)a1.y; a[6]  = (__bf16)a1.z; a[7]  = (__bf16)a1.w;
    a[8] = (__bf16)a2.x; a[9] = (__bf16)a2.y; a[10] = (__bf16)a2.z; a[11] = (__bf16)a2.w;
    a[12]= (__bf16)a3.x; a[13]= (__bf16)a3.y; a[14] = (__bf16)a3.z; a[15] = (__bf16)a3.w;
    afrag[ks] = a;
  }

  bool fullTile = (rbase + 16 <= n);          // wave-uniform predicate
  bool anyRow   = (rbase < n);

#pragma unroll
  for (int tt = 0; tt < 8; ++tt) {
    int sel = tt >> 2;
    int t = tt & 3;
    int ncol = t * 16 + l;
    v8f c = {};
#pragma unroll
    for (int ks = 0; ks < 2; ++ks) {
      // B (32x16 bf16): lane = column, elems 0..7 = K ks*32+8*half.., elems 8..15 = +16 (table order)
      const __bf16* wb = &sW[sel * 4096 + ncol * 64 + ks * 32 + 8 * half];
      v8bf blo = *(const v8bf*)(wb);
      v8bf bhi = *(const v8bf*)(wb + 16);
      v16bf b;
#pragma unroll
      for (int j = 0; j < 8; ++j) { b[j] = blo[j]; b[j + 8] = bhi[j]; }
      c = __builtin_amdgcn_wmma_f32_16x16x32_bf16(false, afrag[ks], false, b,
                                                  (short)0, c, false, false);
    }
    float biasv = (sel ? br : bl)[ncol];
    float* dstp = (sel ? xr : xl) + (size_t)(rbase + 8 * half) * 64 + ncol;
    if (fullTile) {
      // fast path: straight store clause, no per-element EXEC juggling
#pragma unroll
      for (int r = 0; r < 8; ++r)             // C: VGPR r -> M = r + 8*half
        dstp[(size_t)r * 64] = c[r] + biasv;
    } else if (anyRow) {
#pragma unroll
      for (int r = 0; r < 8; ++r) {
        int m = rbase + r + 8 * half;
        if (m < n) dstp[(size_t)r * 64] = c[r] + biasv;
      }
    }
  }
}

// ---------------- kernel 2: edge GEMM + attention logits + segment max ------
// 16 edges per wave.  e = edge_attr @ W_e via WMMA (K=16 zero-padded to 32).
// Column tile t == head t, so logits reduce over the 16 lanes of a half-wave.
__global__ __launch_bounds__(TPB) void k2_edge_logits(
    const float* __restrict__ ea, const float* __restrict__ We,
    const int* __restrict__ src, const int* __restrict__ dst,
    const float* __restrict__ xl, const float* __restrict__ xr,
    const float* __restrict__ att,
    float* __restrict__ logits, unsigned* __restrict__ maxb, int E) {
  __shared__ __attribute__((aligned(16))) __bf16 sWe[1024]; // transposed [col][k], 64x16
  int tid = threadIdx.x;
  for (int idx = tid; idx < 1024; idx += TPB) {
    int ncol = idx >> 4, k = idx & 15;
    sWe[idx] = (__bf16)We[k * 64 + ncol];
  }
  __syncthreads();

  int wave = tid >> 5, lane = tid & 31;
  int half = lane >> 4, l = lane & 15;
  int ebase = (blockIdx.x * 8 + wave) * 16;

  int rowe = ebase + l; if (rowe >= E) rowe = E - 1;
  const float* ep = ea + (size_t)rowe * 16;
  int s1 = 8 * half;                         // real K only 0..15
  float4 a0 = *(const float4*)(ep + s1);
  float4 a1 = *(const float4*)(ep + s1 + 4);
  v16bf a = {};                              // elems 8..15 (K>=16) stay zero
  a[0] = (__bf16)a0.x; a[1] = (__bf16)a0.y; a[2] = (__bf16)a0.z; a[3] = (__bf16)a0.w;
  a[4] = (__bf16)a1.x; a[5] = (__bf16)a1.y; a[6] = (__bf16)a1.z; a[7] = (__bf16)a1.w;

  v8f c[4];
#pragma unroll
  for (int t = 0; t < 4; ++t) {
    int ncol = t * 16 + l;
    v16bf b = {};                            // elems 8..15 (K>=16) -> zero pad
    {
      const __bf16* wb = &sWe[ncol * 16 + 8 * half];
      v8bf blo = *(const v8bf*)(wb);
#pragma unroll
      for (int j = 0; j < 8; ++j) { b[j] = blo[j]; }
    }
    v8f z = {};
    c[t] = __builtin_amdgcn_wmma_f32_16x16x32_bf16(false, a, false, b,
                                                   (short)0, z, false, false);
  }

  // logits: leaky_relu(e + xl[src] + xr[dst]) . att[head], head == tile t
#pragma unroll
  for (int t = 0; t < 4; ++t) {
    int ncol = t * 16 + l;
    float av = att[t * 16 + l];
#pragma unroll
    for (int r = 0; r < 8; ++r) {
      int e = ebase + r + 8 * half;
      int ec = e < E ? e : (E - 1);
      int se = src[ec], de = dst[ec];
      float g = c[t][r] + xl[(size_t)se * 64 + ncol] + xr[(size_t)de * 64 + ncol];
      g = fmaxf(g, 0.0f) + 0.2f * fminf(g, 0.0f);   // leaky relu
      float p = g * av;
      for (int mm = 8; mm >= 1; mm >>= 1) p += __shfl_xor(p, mm, 16);
      if (l == 0 && e < E) {
        logits[(size_t)e * 4 + t] = p;
        atomicMax(&maxb[(size_t)de * 4 + t], ordf(p));
      }
    }
  }
}

// ---------------- kernel 3: exp(logit - max[dst]) and denom -----------------
__global__ __launch_bounds__(TPB) void k3_expsum(
    float* __restrict__ exlog, const int* __restrict__ dst,
    const unsigned* __restrict__ maxb, float* __restrict__ denom, int E) {
  int gid = blockIdx.x * TPB + threadIdx.x;
  if (gid >= E * 4) return;
  int e = gid >> 2, h = gid & 3;
  int d = dst[e];
  float mx = unordf(maxb[(size_t)d * 4 + h]);
  float ex = __expf(exlog[gid] - mx);
  exlog[gid] = ex;
  atomicAdd(&denom[(size_t)d * 4 + h], ex);
}

// ---------------- kernel 4: alpha-weighted scatter aggregation --------------
__global__ __launch_bounds__(TPB) void k4_aggregate(
    const float* __restrict__ exlog, const float* __restrict__ denom,
    const int* __restrict__ src, const int* __restrict__ dst,
    const float* __restrict__ xl, float* __restrict__ agg, int E) {
  long long gid = (long long)blockIdx.x * TPB + threadIdx.x;
  if (gid >= (long long)E * 64) return;
  int e = (int)(gid >> 6), nch = (int)(gid & 63);
  int h = nch >> 4;
  int d = dst[e], s = src[e];
  float alpha = exlog[(size_t)e * 4 + h] / (denom[(size_t)d * 4 + h] + 1e-16f);
  atomicAdd(&agg[(size_t)d * 64 + nch], xl[(size_t)s * 64 + nch] * alpha);
}

// ---------------- kernel 5: BatchNorm statistics ----------------------------
__global__ __launch_bounds__(TPB) void k5_stats(
    const float* __restrict__ agg, const float* __restrict__ bias,
    float* __restrict__ stats, int n) {
  __shared__ float ssum[TPB], ssq[TPB];
  int tid = threadIdx.x;
  int gid = blockIdx.x * TPB + tid;
  int stride = gridDim.x * TPB;               // multiple of 64 -> channel fixed
  int c = gid & 63;
  float sum = 0.0f, sq = 0.0f;
  for (long long idx = gid; idx < (long long)n * 64; idx += stride) {
    float v = agg[idx] + bias[c];
    sum += v; sq += v * v;
  }
  ssum[tid] = sum; ssq[tid] = sq;
  __syncthreads();
  if (tid < 64) {
    float ts = ssum[tid] + ssum[tid + 64] + ssum[tid + 128] + ssum[tid + 192];
    float tq = ssq[tid]  + ssq[tid + 64]  + ssq[tid + 128]  + ssq[tid + 192];
    atomicAdd(&stats[tid], ts);
    atomicAdd(&stats[64 + tid], tq);
  }
}

// ---------------- kernel 6: BN + residual + exact GELU ----------------------
__global__ __launch_bounds__(TPB) void k6_final(
    const float* __restrict__ agg, const float* __restrict__ x,
    const float* __restrict__ bias, const float* __restrict__ gamma,
    const float* __restrict__ beta, const float* __restrict__ stats,
    float* __restrict__ out, int n) {
  int gid = blockIdx.x * TPB + threadIdx.x;
  if (gid >= n * 64) return;
  int c = gid & 63;
  float invn = 1.0f / (float)n;
  float mean = stats[c] * invn;
  float var = stats[64 + c] * invn - mean * mean;
  float rstd = rsqrtf(var + 1e-5f);
  float v = (agg[gid] + bias[c] - mean) * rstd * gamma[c] + beta[c] + x[gid];
  out[gid] = 0.5f * v * (1.0f + erff(v * 0.70710678118654752f));
}

// ---------------------------------------------------------------------------
extern "C" void kernel_launch(void* const* d_in, const int* in_sizes, int n_in,
                              void* d_out, int out_size, void* d_ws, size_t ws_size,
                              hipStream_t stream) {
  const float* x     = (const float*)d_in[0];
  const int*   ei    = (const int*)  d_in[1];
  const float* ea    = (const float*)d_in[2];
  const float* Wl    = (const float*)d_in[3];
  const float* bl    = (const float*)d_in[4];
  const float* Wr    = (const float*)d_in[5];
  const float* br    = (const float*)d_in[6];
  const float* We    = (const float*)d_in[7];
  const float* att   = (const float*)d_in[8];
  const float* bias  = (const float*)d_in[9];
  const float* gamma = (const float*)d_in[10];
  const float* beta  = (const float*)d_in[11];
  int n = in_sizes[0] / 64;        // 50000
  int E = in_sizes[2] / 16;        // 800000
  const int* src = ei;
  const int* dst = ei + E;

  float*    xl    = (float*)d_ws;
  float*    xr    = xl + (size_t)n * 64;
  float*    exlog = xr + (size_t)n * 64;
  unsigned* maxb  = (unsigned*)(exlog + (size_t)E * 4);
  float*    denom = (float*)(maxb + (size_t)n * 4);
  float*    stats = denom + (size_t)n * 4;
  float*    out   = (float*)d_out;

  int gInit = (n * 64 + TPB - 1) / TPB;
  k0_init<<<gInit, TPB, 0, stream>>>(out, maxb, denom, stats, n);

  k1_node_xform<<<(n + 127) / 128, TPB, 0, stream>>>(x, Wl, bl, Wr, br, xl, xr, n);

  k2_edge_logits<<<(E + 127) / 128, TPB, 0, stream>>>(ea, We, src, dst, xl, xr,
                                                      att, exlog, maxb, E);

  k3_expsum<<<(E * 4 + TPB - 1) / TPB, TPB, 0, stream>>>(exlog, dst, maxb, denom, E);

  long long tot4 = (long long)E * 64;
  k4_aggregate<<<(int)((tot4 + TPB - 1) / TPB), TPB, 0, stream>>>(exlog, denom,
                                                                  src, dst, xl, out, E);

  k5_stats<<<256, TPB, 0, stream>>>(out, bias, stats, n);

  k6_final<<<gInit, TPB, 0, stream>>>(out, x, bias, gamma, beta, stats, out, n);
}
